// VisionMamba_8641474200046
// MI455X (gfx1250) — matrix operand order repaired and checked by reading, not compiled, not run on hardware
//
#include <hip/hip_runtime.h>
#include <hip/hip_bf16.h>

#define BATCH 8
#define SEQ   4096
#define DIM   768
#define DIM2  1536
#define MROWS (BATCH * SEQ)   // 32768

typedef __attribute__((ext_vector_type(16))) __bf16 v16bf;
typedef __attribute__((ext_vector_type(8)))  float  v8f;
typedef unsigned short u16;
typedef unsigned int   u32;

// ---------- helpers ----------
__device__ __forceinline__ u16 f2bf(float f) {
    u32 u = __float_as_uint(f);
    u32 r = u + 0x7FFFu + ((u >> 16) & 1u);   // round-to-nearest-even
    return (u16)(r >> 16);
}
__device__ __forceinline__ float sigmoidf_(float x) { return 1.0f / (1.0f + __expf(-x)); }

// Low 32 bits of a flat shared-memory address are the LDS byte offset
// (ISA: LDS aperture base occupies addr[63:32]).
__device__ __forceinline__ unsigned lds_off_u32(const void* p) {
    return (unsigned)(unsigned long long)p;
}

// ---------- weight packing (f32 -> bf16, transposed to [N][K]) ----------
__global__ void pack_wi(const float* __restrict__ Wi, u16* __restrict__ WiT) {
    int i = blockIdx.x * blockDim.x + threadIdx.x;       // over N*K = 1536*768
    if (i >= DIM2 * DIM) return;
    int n = i / DIM, k = i % DIM;
    WiT[i] = f2bf(Wi[(size_t)k * DIM2 + n]);
}
__global__ void pack_wc(const float* __restrict__ Wf, const float* __restrict__ Wb,
                        u16* __restrict__ WcT) {
    int i = blockIdx.x * blockDim.x + threadIdx.x;       // over 3*1536*768
    if (i >= 3 * DIM2 * DIM) return;
    int t = i / (DIM2 * DIM);
    int r = i % (DIM2 * DIM);
    int n = r / DIM, k = r % DIM;                        // n = out channel (2D), k = in channel
    float v;
    if (n < DIM) v = Wf[(size_t)n * DIM * 3 + (size_t)k * 3 + t];               // forward tap t
    else         v = Wb[(size_t)(n - DIM) * DIM * 3 + (size_t)k * 3 + (2 - t)]; // reversed taps
    WcT[i] = f2bf(v);
}
__global__ void pack_wo(const float* __restrict__ Wo, u16* __restrict__ WoT) {
    int i = blockIdx.x * blockDim.x + threadIdx.x;       // over 768*768
    if (i >= DIM * DIM) return;
    int n = i / DIM, k = i % DIM;
    WoT[i] = f2bf(Wo[(size_t)k * DIM + n]);
}
__global__ void pack_biasc(const float* __restrict__ bfc, const float* __restrict__ bbc,
                           float* __restrict__ biasc) {
    int i = blockIdx.x * blockDim.x + threadIdx.x;
    if (i >= DIM2) return;
    biasc[i] = (i < DIM) ? bfc[i] : bbc[i - DIM];
}

// ---------- LayerNorm -> bf16 ----------
__global__ void __launch_bounds__(256) ln_kernel(const float* __restrict__ x,
                                                 const float* __restrict__ g,
                                                 const float* __restrict__ b,
                                                 u16* __restrict__ xn) {
    __shared__ float red[256];
    int row = blockIdx.x;                                 // 0..MROWS-1
    const float* xr = x + (size_t)row * DIM;
    int t = threadIdx.x;
    float v0 = xr[t], v1 = xr[t + 256], v2 = xr[t + 512];
    red[t] = v0 + v1 + v2;
    __syncthreads();
    for (int s = 128; s > 0; s >>= 1) { if (t < s) red[t] += red[t + s]; __syncthreads(); }
    float mu = red[0] * (1.0f / DIM);
    __syncthreads();
    float d0 = v0 - mu, d1 = v1 - mu, d2 = v2 - mu;
    red[t] = d0 * d0 + d1 * d1 + d2 * d2;
    __syncthreads();
    for (int s = 128; s > 0; s >>= 1) { if (t < s) red[t] += red[t + s]; __syncthreads(); }
    float rs = __frsqrt_rn(red[0] * (1.0f / DIM) + 1e-5f);
    u16* xo = xn + (size_t)row * DIM;
    xo[t]       = f2bf(d0 * rs * g[t]       + b[t]);
    xo[t + 256] = f2bf(d1 * rs * g[t + 256] + b[t + 256]);
    xo[t + 512] = f2bf(d2 * rs * g[t + 512] + b[t + 512]);
}

// ---------- generic bf16 WMMA GEMM: C[M,N] = sum_t shiftA(t) @ B_t + bias (+res) ----------
// A: bf16 [M x K] row-major. Bp: bf16 [taps][N][K] (B transposed). C: f32 [M x N].
// For taps==3, tap t uses A rows shifted by (t-1) within each length-L sequence (zero pad).
// Double-buffered LDS; tiles filled with GLOBAL_LOAD_ASYNC_TO_LDS_B128 (ASYNCcnt),
// next stage issued while current stage runs through the WMMA pipe.
__global__ void __launch_bounds__(256) gemm_wmma(const u16* __restrict__ A,
                                                 const u16* __restrict__ Bp,
                                                 float* __restrict__ C,
                                                 int M, int N, int K, int taps, int L,
                                                 const float* __restrict__ bias,
                                                 const float* __restrict__ residual) {
    constexpr int LDA = 48;                        // halves; rows 16B aligned
    constexpr int TILE = 128 * LDA;                // halves per tile buffer
    __shared__ u16 As[2][TILE];
    __shared__ u16 Bs[2][TILE];

    const int tid  = threadIdx.x;
    const int wave = tid >> 5, lane = tid & 31;
    const int half = lane >> 4, hidx = lane & 15;
    const int wm = wave >> 2, wn = wave & 3;       // 2x4 wave grid
    const int bm = blockIdx.y * 128, bn = blockIdx.x * 128;

    const int r0   = tid >> 2;                     // this thread's tile row (chunk 0)
    const int col0 = (tid & 3) * 8;                // halves, 16B granules
    const int lbase = bm % L;                      // 128 | L: whole tile in one sequence
    const int ksteps  = K >> 5;
    const int nstages = taps * ksteps;

    const unsigned aBase = lds_off_u32(&As[0][0]);
    const unsigned bBase = lds_off_u32(&Bs[0][0]);
    const unsigned tileB = (unsigned)TILE * 2u;    // bytes per buffer

    union FragA { v16bf v; u32 u[8]; };
    union FragB { v16bf v; u32 u[8]; };
    v8f acc[4][2] = {};

    auto issue = [&](int s, int buf) {
        int t  = s / ksteps;
        int k0 = (s - t * ksteps) << 5;
        int shift = t - (taps >> 1);
        const u16* Bt = Bp + (size_t)t * N * K;
#pragma unroll
        for (int c = 0; c < 2; ++c) {
            int r  = r0 + c * 64;                  // 0..127
            int gr = bm + r;
            unsigned aoff = aBase + (unsigned)buf * tileB + (unsigned)(r * LDA + col0) * 2u;
            int lsrc = lbase + r + shift;
            if (lsrc >= 0 && lsrc < L) {
                const u16* gp = A + (size_t)(gr + shift) * K + k0 + col0;
                asm volatile("global_load_async_to_lds_b128 %0, %1, off"
                             :: "v"(aoff), "v"(gp) : "memory");
            } else {
                uint4 z = {0u, 0u, 0u, 0u};        // zero-pad at sequence boundary
                *(uint4*)&As[buf][r * LDA + col0] = z;
            }
            unsigned boff = bBase + (unsigned)buf * tileB + (unsigned)(r * LDA + col0) * 2u;
            const u16* gbp = Bt + (size_t)(bn + r) * K + k0 + col0;
            asm volatile("global_load_async_to_lds_b128 %0, %1, off"
                         :: "v"(boff), "v"(gbp) : "memory");
        }
    };

    issue(0, 0);
    for (int s = 0; s < nstages; ++s) {
        int buf = s & 1;
        // each wave drains its own async loads, then the barrier makes completion
        // workgroup-wide (and orders the zero-fill ds_stores + last stage's reads)
        asm volatile("s_wait_asynccnt 0x0" ::: "memory");
        __syncthreads();
        if (s + 1 < nstages) issue(s + 1, buf ^ 1);

        const u16* as = As[buf];
        const u16* bs = Bs[buf];
        FragA af[4];
        FragB bf2[2];
#pragma unroll
        for (int i = 0; i < 4; ++i) {
            const u16* p = as + (wm * 64 + i * 16 + hidx) * LDA;
#pragma unroll
            for (int d = 0; d < 8; ++d) {
                int kk = (d < 4) ? (half * 8 + 2 * d) : (16 + half * 8 + 2 * (d - 4));
                af[i].u[d] = *(const u32*)(p + kk);
            }
        }
#pragma unroll
        for (int j = 0; j < 2; ++j) {
            const u16* p = bs + (wn * 32 + j * 16 + hidx) * LDA;
#pragma unroll
            for (int d = 0; d < 8; ++d) {
                int kk = (d < 4) ? (half * 8 + 2 * d) : (16 + half * 8 + 2 * (d - 4));        // B frag: same K order as A frag
                bf2[j].u[d] = *(const u32*)(p + kk);
            }
        }
#pragma unroll
        for (int i = 0; i < 4; ++i)
#pragma unroll
            for (int j = 0; j < 2; ++j)
                acc[i][j] = __builtin_amdgcn_wmma_f32_16x16x32_bf16(
                    false, af[i].v, false, bf2[j].v, (short)0, acc[i][j], false, false);
    }

    // epilogue: C element (m,n) lives at vgpr v=m%8, lane (m/8)*16+n
#pragma unroll
    for (int i = 0; i < 4; ++i) {
#pragma unroll
        for (int j = 0; j < 2; ++j) {
            int m0 = bm + wm * 64 + i * 16 + half * 8;
            int n  = bn + wn * 32 + j * 16 + hidx;
            float badd = bias ? bias[n] : 0.0f;
#pragma unroll
            for (int v = 0; v < 8; ++v) {
                int m = m0 + v;
                float val = acc[i][j][v] + badd;
                if (residual) val += residual[(size_t)m * N + n];
                C[(size_t)m * N + n] = val;
            }
        }
    }
}

// ---------- split proj -> xg(bf16) + silu(z)(f32) ----------
__global__ void split_kernel(const float* __restrict__ proj, u16* __restrict__ xg,
                             float* __restrict__ zg) {
    int i = blockIdx.x * blockDim.x + threadIdx.x;
    if (i >= MROWS * DIM) return;
    int m = i / DIM, d = i % DIM;
    float p = proj[(size_t)m * DIM2 + d];
    float z = proj[(size_t)m * DIM2 + DIM + d];
    xg[i] = f2bf(p);
    zg[i] = z * sigmoidf_(z);
}

// ---------- in-place sequential SSM scans (fwd on cols [0,D), bwd on cols [D,2D)) ----------
__global__ void __launch_bounds__(256) ssm_kernel(float* __restrict__ pre,
        const float* __restrict__ af, const float* __restrict__ bfp,
        const float* __restrict__ cf, const float* __restrict__ df,
        const float* __restrict__ ab, const float* __restrict__ bbp,
        const float* __restrict__ cb, const float* __restrict__ db) {
    int tid = blockIdx.x * blockDim.x + threadIdx.x;     // 0 .. 2*B*D-1
    if (tid >= 2 * BATCH * DIM) return;
    int dir = tid / (BATCH * DIM);
    int rem = tid % (BATCH * DIM);
    int b = rem / DIM, d = rem % DIM;
    float a, bp, cp, dp;
    int col;
    if (dir == 0) { a = sigmoidf_(af[d]); bp = bfp[d]; cp = cf[d]; dp = df[d]; col = d; }
    else          { a = sigmoidf_(ab[d]); bp = bbp[d]; cp = cb[d]; dp = db[d]; col = DIM + d; }
    size_t base = (size_t)b * SEQ * DIM2 + col;
    float h = 0.0f;
    for (int i = 0; i < SEQ; ++i) {
        int l = dir ? (SEQ - 1 - i) : i;                 // bwd dir scans reverse time
        size_t idx = base + (size_t)l * DIM2;
        float u = pre[idx];
        h = fmaf(a, h, bp * u);
        pre[idx] = fmaf(cp, h, dp * u);
    }
}

// ---------- combine: ycomb = bf16((y_f + y_b) * silu(z)) ----------
__global__ void combine_kernel(const float* __restrict__ pre, const float* __restrict__ zg,
                               u16* __restrict__ ycomb) {
    int i = blockIdx.x * blockDim.x + threadIdx.x;
    if (i >= MROWS * DIM) return;
    int m = i / DIM, d = i % DIM;
    float yf = pre[(size_t)m * DIM2 + d];
    float yb = pre[(size_t)m * DIM2 + DIM + d];
    ycomb[i] = f2bf((yf + yb) * zg[i]);
}

// ---------- launch ----------
extern "C" void kernel_launch(void* const* d_in, const int* in_sizes, int n_in,
                              void* d_out, int out_size, void* d_ws, size_t ws_size,
                              hipStream_t stream) {
    const float* x    = (const float*)d_in[0];
    const float* ln_g = (const float*)d_in[1];
    const float* ln_b = (const float*)d_in[2];
    const float* Wi   = (const float*)d_in[3];
    const float* bi   = (const float*)d_in[4];
    const float* Wf   = (const float*)d_in[5];
    const float* bfc  = (const float*)d_in[6];
    const float* af   = (const float*)d_in[7];
    const float* bf_  = (const float*)d_in[8];
    const float* cf   = (const float*)d_in[9];
    const float* df   = (const float*)d_in[10];
    const float* Wb   = (const float*)d_in[11];
    const float* bbc  = (const float*)d_in[12];
    const float* ab   = (const float*)d_in[13];
    const float* bb_  = (const float*)d_in[14];
    const float* cb   = (const float*)d_in[15];
    const float* db   = (const float*)d_in[16];
    const float* Wo   = (const float*)d_in[17];
    const float* bo   = (const float*)d_in[18];
    float* out = (float*)d_out;

    char* base = (char*)d_ws;
    size_t off = 0;
    auto take = [&](size_t bytes) { char* p = base + off; off += (bytes + 255) & ~(size_t)255; return p; };

    u16*   xn    = (u16*)  take((size_t)MROWS * DIM * 2);      // aliased later by ycomb
    float* proj  = (float*)take((size_t)MROWS * DIM2 * 4);     // aliased by convpre
    u16*   xg    = (u16*)  take((size_t)MROWS * DIM * 2);
    float* zg    = (float*)take((size_t)MROWS * DIM * 4);
    u16*   WiT   = (u16*)  take((size_t)DIM2 * DIM * 2);
    u16*   WcT   = (u16*)  take((size_t)3 * DIM2 * DIM * 2);
    u16*   WoT   = (u16*)  take((size_t)DIM * DIM * 2);
    float* biasc = (float*)take((size_t)DIM2 * 4);
    u16*   ycomb = xn;            // xn dead after proj GEMM
    float* convpre = proj;        // proj dead after split

    // 1) pack weights to bf16 [N][K]
    pack_wi<<<(DIM2 * DIM + 255) / 256, 256, 0, stream>>>(Wi, WiT);
    pack_wc<<<(3 * DIM2 * DIM + 255) / 256, 256, 0, stream>>>(Wf, Wb, WcT);
    pack_wo<<<(DIM * DIM + 255) / 256, 256, 0, stream>>>(Wo, WoT);
    pack_biasc<<<(DIM2 + 255) / 256, 256, 0, stream>>>(bfc, bbc, biasc);

    // 2) layernorm -> bf16
    ln_kernel<<<MROWS, 256, 0, stream>>>(x, ln_g, ln_b, xn);

    // 3) proj GEMM: [32768x768] @ [768x1536] + bi
    gemm_wmma<<<dim3(DIM2 / 128, MROWS / 128), 256, 0, stream>>>(
        xn, WiT, proj, MROWS, DIM2, DIM, 1, SEQ, bi, nullptr);

    // 4) split + SiLU
    split_kernel<<<(MROWS * DIM + 255) / 256, 256, 0, stream>>>(proj, xg, zg);

    // 5) both convs as one 3-tap shifted GEMM: [32768x768] @ 3x[768x1536]
    gemm_wmma<<<dim3(DIM2 / 128, MROWS / 128), 256, 0, stream>>>(
        xg, WcT, convpre, MROWS, DIM2, DIM, 3, SEQ, biasc, nullptr);

    // 6) SSM scans in place
    ssm_kernel<<<(2 * BATCH * DIM + 255) / 256, 256, 0, stream>>>(
        convpre, af, bf_, cf, df, ab, bb_, cb, db);

    // 7) gated combine -> bf16
    combine_kernel<<<(MROWS * DIM + 255) / 256, 256, 0, stream>>>(convpre, zg, ycomb);

    // 8) output GEMM + bias + residual
    gemm_wmma<<<dim3(DIM / 128, MROWS / 128), 256, 0, stream>>>(
        ycomb, WoT, out, MROWS, DIM, DIM, 1, SEQ, bo, x);
}
